// PointNetSetAbstraction_12103217840690
// MI455X (gfx1250) — matrix operand order repaired and checked by reading, not compiled, not run on hardware
//
#include <hip/hip_runtime.h>
#include <math.h>

typedef __attribute__((ext_vector_type(16))) _Float16 v16h;
typedef __attribute__((ext_vector_type(8)))  float    v8f;
typedef __attribute__((ext_vector_type(4)))  int      v4i;

#define B_ 16
#define N_ 8192
#define S_ 1024
#define K_ 32
#define M_TOTAL (B_*S_*K_)   // 524288

#if defined(__has_builtin)
#if __has_builtin(__builtin_amdgcn_global_load_async_to_lds_b128)
#define HAVE_ASYNC_LDS 1
#endif
#endif

#ifdef HAVE_ASYNC_LDS
typedef __attribute__((address_space(1))) v4i gbl_v4i;
typedef __attribute__((address_space(3))) v4i lds_v4i;
#endif

// =====================================================================
// 1. Farthest point sampling: one block per batch, whole cloud in LDS
//    (128 KB of the 320 KB WGP LDS). Writes new_xyz to d_out[0:B*S*3].
// =====================================================================
__global__ __launch_bounds__(1024) void fps_kernel(const float* __restrict__ xyz,
                                                   float* __restrict__ newXYZ)
{
    extern __shared__ char smem_raw[];
    float* xs   = (float*)smem_raw;
    float* ys   = xs + N_;
    float* zs   = ys + N_;
    float* dist = zs + N_;
    float* redv = dist + N_;
    int*   redi = (int*)(redv + 32);
    int*   slot = redi + 32;

    const int b   = blockIdx.x;
    const int tid = threadIdx.x;
    const float* base = xyz + (size_t)b * N_ * 3;

    for (int i = tid; i < N_ * 3; i += 1024) {
        float v = base[i];
        int n = i / 3, c = i - n * 3;
        if (c == 0) xs[n] = v; else if (c == 1) ys[n] = v; else zs[n] = v;
    }
    for (int n = tid; n < N_; n += 1024) dist[n] = 1e10f;
    __syncthreads();

    int far = 0;
    for (int s = 0; s < S_; ++s) {
        float cx = xs[far], cy = ys[far], cz = zs[far];
        if (tid == 0) {
            float* o = newXYZ + ((size_t)b * S_ + s) * 3;
            o[0] = cx; o[1] = cy; o[2] = cz;
        }
        float bestv = -1.0f; int besti = 0;
        #pragma unroll
        for (int j = 0; j < N_ / 1024; ++j) {
            int n = tid + j * 1024;
            float dx = xs[n] - cx, dy = ys[n] - cy, dz = zs[n] - cz;
            float d  = dx*dx + dy*dy + dz*dz;
            float nd = fminf(dist[n], d);
            dist[n] = nd;
            if (nd > bestv) { bestv = nd; besti = n; }
        }
        #pragma unroll
        for (int off = 16; off > 0; off >>= 1) {
            float ov = __shfl_down(bestv, off, 32);
            int   oi = __shfl_down(besti, off, 32);
            if (ov > bestv || (ov == bestv && oi < besti)) { bestv = ov; besti = oi; }
        }
        int wid = tid >> 5;
        if ((tid & 31) == 0) { redv[wid] = bestv; redi[wid] = besti; }
        __syncthreads();
        if (tid < 32) {
            bestv = redv[tid]; besti = redi[tid];
            #pragma unroll
            for (int off = 16; off > 0; off >>= 1) {
                float ov = __shfl_down(bestv, off, 32);
                int   oi = __shfl_down(besti, off, 32);
                if (ov > bestv || (ov == bestv && oi < besti)) { bestv = ov; besti = oi; }
            }
            if (tid == 0) *slot = besti;
        }
        __syncthreads();
        far = *slot;
    }
}

// =====================================================================
// 2. KNN (32 nearest of 8192) fused with gather + concat + f16 pack.
//    Emits X0 rows [M_TOTAL, 16] f16: (dxyz, 6 feats, 7 zero pad).
// =====================================================================
__global__ __launch_bounds__(256) void knn_group_kernel(const float* __restrict__ xyz,
                                                        const float* __restrict__ pts,
                                                        const float* __restrict__ newXYZ,
                                                        _Float16* __restrict__ X0)
{
    __shared__ float lx[1024], ly[1024], lz[1024];
    const int tid = threadIdx.x;
    const int q   = blockIdx.x * 256 + tid;   // 0 .. B*S-1
    const int b   = q >> 10;
    const float* base = xyz + (size_t)b * N_ * 3;

    float qx = newXYZ[(size_t)q * 3 + 0];
    float qy = newXYZ[(size_t)q * 3 + 1];
    float qz = newXYZ[(size_t)q * 3 + 2];

    float bd[K_]; int bi[K_];
    #pragma unroll
    for (int k = 0; k < K_; ++k) { bd[k] = 3e38f; bi[k] = 0; }
    float wv = 3e38f; int ws = 0;

    for (int c0 = 0; c0 < N_; c0 += 1024) {
        __syncthreads();
        for (int i = tid; i < 1024 * 3; i += 256) {
            float v = base[(size_t)c0 * 3 + i];
            int n = i / 3, c = i - n * 3;
            if (c == 0) lx[n] = v; else if (c == 1) ly[n] = v; else lz[n] = v;
        }
        __syncthreads();
        for (int n = 0; n < 1024; ++n) {
            float dx = qx - lx[n], dy = qy - ly[n], dz = qz - lz[n];
            float d = dx*dx + dy*dy + dz*dz;
            if (d < wv) {
                bd[ws] = d; bi[ws] = c0 + n;
                wv = -1.0f;
                #pragma unroll
                for (int t = 0; t < K_; ++t)
                    if (bd[t] > wv) { wv = bd[t]; ws = t; }
            }
        }
    }

    const float* pbase = pts + (size_t)b * N_ * 6;
    for (int k = 0; k < K_; ++k) {
        int n = bi[k];
        _Float16* row = X0 + ((size_t)q * K_ + k) * 16;
        row[0] = (_Float16)(base[(size_t)n*3+0] - qx);
        row[1] = (_Float16)(base[(size_t)n*3+1] - qy);
        row[2] = (_Float16)(base[(size_t)n*3+2] - qz);
        #pragma unroll
        for (int c = 0; c < 6; ++c)
            row[3 + c] = (_Float16)pbase[(size_t)n*6 + c];
        #pragma unroll
        for (int c = 9; c < 16; ++c)
            row[c] = (_Float16)0.0f;
    }
}

// =====================================================================
// 3. WMMA GEMM: Y(f16) = act(Xprev) @ W^T + bias, plus per-channel f32
//    sum/sumsq for training-mode BN. One 16-row M tile staged in LDS
//    per step; COUT/16 waves each own a 16-column N tile. Layer-0 A
//    staging uses the CDNA5 async global->LDS copy path when available.
// =====================================================================
template<int CIN, int COUT, bool FIRST, int MT>
__global__ void gemm_kernel(const _Float16* __restrict__ X0,
                            const _Float16* __restrict__ Yprev,
                            const float* __restrict__ W,
                            const float* __restrict__ bias,
                            const float* __restrict__ bnScale,
                            const float* __restrict__ bnShift,
                            _Float16* __restrict__ Yout,
                            float* __restrict__ sumOut,
                            float* __restrict__ sumsqOut)
{
    static_assert(CIN % 32 == 0 && COUT % 16 == 0, "tile shapes");
    constexpr int KCH  = CIN / 32;
    constexpr int NW   = COUT / 16;
    constexpr int CINR = FIRST ? 9 : CIN;   // real (unpadded) K extent

    __shared__ _Float16 ldsA[16 * CIN];

    const int tid  = threadIdx.x;
    const int lane = tid & 31;
    const int wave = tid >> 5;
    const int h    = lane >> 4;      // half-wave select
    const int cl   = lane & 15;
    const int cg   = wave * 16 + cl; // global output channel

    // B fragment (weights), built once: lane = column, K = 16*h + e (+32q)
    v16h bfrag[KCH];
    #pragma unroll
    for (int q = 0; q < KCH; ++q) {
        #pragma unroll
        for (int e = 0; e < 16; ++e) {
            int K = 32 * q + 16 * h + e;
            float wv = (K < CINR) ? W[cg * CINR + K] : 0.0f;
            bfrag[q][e] = (_Float16)wv;
        }
    }
    const float biasv = bias[cg];
    float sAcc = 0.0f, s2Acc = 0.0f;

    if (FIRST) {
        // K = 16..31 is zero padding; write it once (never overwritten).
        uint32_t* dst = (uint32_t*)ldsA;
        for (int i = tid; i < 16 * (CIN / 2); i += NW * 32)
            if ((i & 15) >= 8) dst[i] = 0u;
        __syncthreads();
    }

    for (int mt = 0; mt < MT; ++mt) {
        const int mbase = (blockIdx.x * MT + mt) * 16;

        if (FIRST) {
            // Stage 16 rows x 16 halves (512B): 32 chunks of 16B.
            if (tid < 32) {
                int r = tid >> 1, d = tid & 1;
                const _Float16* g = X0 + (size_t)(mbase + r) * 16 + d * 8;
                _Float16*       l = ldsA + r * CIN + d * 8;
#ifdef HAVE_ASYNC_LDS
                __builtin_amdgcn_global_load_async_to_lds_b128(
                    (gbl_v4i*)g, (lds_v4i*)l, 0, 0);
#else
                const uint32_t* gs = (const uint32_t*)g;
                uint32_t*       ls = (uint32_t*)l;
                ls[0] = gs[0]; ls[1] = gs[1]; ls[2] = gs[2]; ls[3] = gs[3];
#endif
            }
#ifdef HAVE_ASYNC_LDS
            asm volatile("s_wait_asynccnt 0x0" ::: "memory");
#endif
        } else {
            // Apply previous layer's BN + ReLU while converting to f16.
            for (int i = tid; i < 16 * CIN; i += NW * 32) {
                int r = i / CIN, c = i - r * CIN;
                float y = (float)Yprev[(size_t)(mbase + r) * CIN + c];
                y = fmaxf(y * bnScale[c] + bnShift[c], 0.0f);
                ldsA[i] = (_Float16)y;
            }
            if (mt + 1 < MT)
                __builtin_prefetch(&Yprev[(size_t)(mbase + 16) * CIN], 0, 0);
        }
        __syncthreads();

        v8f acc = {};
        #pragma unroll
        for (int q = 0; q < KCH; ++q) {
            v16h a;
            #pragma unroll
            for (int e = 0; e < 16; ++e) {
                int K = 32 * q + 16 * h + e;
                a[e] = ldsA[cl * CIN + K];
            }
            acc = __builtin_amdgcn_wmma_f32_16x16x32_f16(
                false, a, false, bfrag[q], (short)0, acc, false, false);
        }
        __syncthreads();

        #pragma unroll
        for (int j = 0; j < 8; ++j) {
            int rg  = mbase + h * 8 + j;
            float y = acc[j] + biasv;
            Yout[(size_t)rg * COUT + cg] = (_Float16)y;
            sAcc  += y;
            s2Acc += y * y;
        }
    }

    // lanes c and c+16 hold the same column: combine, then one atomic each.
    sAcc  += __shfl_xor(sAcc, 16, 32);
    s2Acc += __shfl_xor(s2Acc, 16, 32);
    if (h == 0) {
        atomicAdd(&sumOut[cg],   sAcc);
        atomicAdd(&sumsqOut[cg], s2Acc);
    }
}

// =====================================================================
// 4. BN stat finalize: scale = g*rsqrt(var+eps), shift = beta - mean*scale
// =====================================================================
__global__ void bn_finalize_kernel(const float* __restrict__ sum,
                                   const float* __restrict__ sumsq,
                                   const float* __restrict__ g,
                                   const float* __restrict__ beta,
                                   float* __restrict__ scale,
                                   float* __restrict__ shift,
                                   int C, float invM)
{
    int c = threadIdx.x;
    if (c < C) {
        float mean = sum[c] * invM;
        float var  = sumsq[c] * invM - mean * mean;
        float sc   = g[c] * rsqrtf(var + 1e-5f);
        scale[c] = sc;
        shift[c] = beta[c] - mean * sc;
    }
}

// =====================================================================
// 5. Final BN + ReLU + max over K -> d_out feature block
// =====================================================================
__global__ __launch_bounds__(256) void bn_relu_maxpool_kernel(const _Float16* __restrict__ Y2,
                                                              const float* __restrict__ scale,
                                                              const float* __restrict__ shift,
                                                              float* __restrict__ out)
{
    int t = blockIdx.x * 256 + threadIdx.x;   // over B*S*128
    int c = t & 127;
    int q = t >> 7;
    float sc = scale[c], sh = shift[c];
    const _Float16* y = Y2 + (size_t)q * K_ * 128 + c;
    float m = -3e38f;
    #pragma unroll 4
    for (int k = 0; k < K_; ++k) {
        float v = fmaxf((float)y[(size_t)k * 128] * sc + sh, 0.0f);
        m = fmaxf(m, v);
    }
    out[t] = m;
}

__global__ void zero_kernel(float* p, int n)
{
    int i = blockIdx.x * 256 + threadIdx.x;
    if (i < n) p[i] = 0.0f;
}

// =====================================================================
// Launcher
// =====================================================================
extern "C" void kernel_launch(void* const* d_in, const int* in_sizes, int n_in,
                              void* d_out, int out_size, void* d_ws, size_t ws_size,
                              hipStream_t stream)
{
    const float* xyz = (const float*)d_in[0];
    const float* pts = (const float*)d_in[1];
    const float* w0  = (const float*)d_in[2];
    const float* b0  = (const float*)d_in[3];
    const float* g0  = (const float*)d_in[4];
    const float* bb0 = (const float*)d_in[5];
    const float* w1  = (const float*)d_in[6];
    const float* b1  = (const float*)d_in[7];
    const float* g1  = (const float*)d_in[8];
    const float* bb1 = (const float*)d_in[9];
    const float* w2  = (const float*)d_in[10];
    const float* b2  = (const float*)d_in[11];
    const float* g2  = (const float*)d_in[12];
    const float* bb2 = (const float*)d_in[13];

    char* ws = (char*)d_ws;
    _Float16* X0   = (_Float16*)ws;                                   // 16 MB
    _Float16* bufA = (_Float16*)(ws + (size_t)16  * 1024 * 1024);     // 128 MB (Y0 f16, then Y2 f16)
    _Float16* bufB = (_Float16*)(ws + (size_t)144 * 1024 * 1024);     // 64 MB  (Y1 f16)
    float*    stats = (float*)  (ws + (size_t)208 * 1024 * 1024);     // 6 KB

    float* sum0 = stats +    0; float* sq0 = stats +  128;
    float* sc0  = stats +  256; float* sh0 = stats +  384;
    float* sum1 = stats +  512; float* sq1 = stats +  640;
    float* sc1  = stats +  768; float* sh1 = stats +  896;
    float* sum2 = stats + 1024; float* sq2 = stats + 1152;
    float* sc2  = stats + 1280; float* sh2 = stats + 1408;

    float* outXYZ  = (float*)d_out;
    float* outFeat = outXYZ + (size_t)B_ * S_ * 3;

    const float invM = 1.0f / (float)M_TOTAL;
    const size_t fpsSmem = (size_t)4 * N_ * sizeof(float) + 32 * sizeof(float) + 33 * sizeof(int);

    zero_kernel<<<(1536 + 255) / 256, 256, 0, stream>>>(stats, 1536);

    fps_kernel<<<B_, 1024, fpsSmem, stream>>>(xyz, outXYZ);

    knn_group_kernel<<<(B_ * S_) / 256, 256, 0, stream>>>(xyz, pts, outXYZ, X0);

    gemm_kernel<32, 64, true, 8><<<M_TOTAL / 128, 128, 0, stream>>>(
        X0, nullptr, w0, b0, nullptr, nullptr, bufA, sum0, sq0);
    bn_finalize_kernel<<<1, 64, 0, stream>>>(sum0, sq0, g0, bb0, sc0, sh0, 64, invM);

    gemm_kernel<64, 64, false, 8><<<M_TOTAL / 128, 128, 0, stream>>>(
        nullptr, bufA, w1, b1, sc0, sh0, bufB, sum1, sq1);
    bn_finalize_kernel<<<1, 64, 0, stream>>>(sum1, sq1, g1, bb1, sc1, sh1, 64, invM);

    gemm_kernel<64, 128, false, 8><<<M_TOTAL / 128, 256, 0, stream>>>(
        nullptr, bufB, w2, b2, sc1, sh1, bufA, sum2, sq2);
    bn_finalize_kernel<<<1, 128, 0, stream>>>(sum2, sq2, g2, bb2, sc2, sh2, 128, invM);

    bn_relu_maxpool_kernel<<<(B_ * S_ * 128) / 256, 256, 0, stream>>>(bufA, sc2, sh2, outFeat);
}
